// CGRModel_38276748542304
// MI455X (gfx1250) — matrix operand order repaired and checked by reading, not compiled, not run on hardware
//
#include <hip/hip_runtime.h>

#define B_   16
#define L_   1024
#define D_   512
#define H_   8
#define DH_  64
#define E_   6
#define NROW (B_*L_)
#define NEGV (-1.0e9f)

typedef unsigned short bf16;
typedef __attribute__((ext_vector_type(16))) __bf16 v16bf;
typedef __attribute__((ext_vector_type(8)))  __bf16 v8bf;
typedef __attribute__((ext_vector_type(8)))  float  v8f;

enum { EP_F32_RES = 0, EP_BF16 = 1, EP_BF16_RELU = 2, EP_QKV = 3 };

static __device__ __forceinline__ bf16 f2bf(float f) {
  unsigned u = __float_as_uint(f);
  unsigned r = u + 0x7FFFu + ((u >> 16) & 1u);
  return (bf16)(r >> 16);
}
static __device__ __forceinline__ float bf2f(bf16 h) {
  return __uint_as_float(((unsigned)h) << 16);
}
// Build a v16bf A-fragment from two contiguous 8-half (16B) chunks.
static __device__ __forceinline__ v16bf load2x8(const bf16* p0, const bf16* p1) {
  v8bf lo = *(const v8bf*)p0;
  v8bf hi = *(const v8bf*)p1;
  v16bf r;
#pragma unroll
  for (int i = 0; i < 8; i++) { r[i] = lo[i]; r[i + 8] = hi[i]; }
  return r;
}
static __device__ __forceinline__ v8f wmma_bf16(v16bf a, v16bf b, v8f c) {
  return __builtin_amdgcn_wmma_f32_16x16x32_bf16(false, a, false, b, (short)0, c, false, false);
}

// ---------------------------------------------------------------------------
// LayerNorm (f32 in) -> bf16 out. One block of 128 threads per row of D_=512.
// ---------------------------------------------------------------------------
__global__ __launch_bounds__(128)
void ln_bf16_kernel(const float* __restrict__ x, const float* __restrict__ g,
                    const float* __restrict__ bta, bf16* __restrict__ out) {
  __shared__ float red[128];
  const int row = blockIdx.x;
  const int t = threadIdx.x;
  const float* xr = x + (size_t)row * D_;
  float4 v = *(const float4*)(xr + t * 4);
  float s = v.x + v.y + v.z + v.w;
  float s2 = v.x * v.x + v.y * v.y + v.z * v.z + v.w * v.w;
  red[t] = s; __syncthreads();
  for (int o = 64; o > 0; o >>= 1) { if (t < o) red[t] += red[t + o]; __syncthreads(); }
  const float mu = red[0] * (1.0f / D_);
  __syncthreads();
  red[t] = s2; __syncthreads();
  for (int o = 64; o > 0; o >>= 1) { if (t < o) red[t] += red[t + o]; __syncthreads(); }
  const float var = red[0] * (1.0f / D_) - mu * mu;
  const float rstd = rsqrtf(var + 1e-5f);
  bf16* orow = out + (size_t)row * D_;
  const float* vv = (const float*)&v;
#pragma unroll
  for (int i = 0; i < 4; i++) {
    const int d = t * 4 + i;
    orow[d] = f2bf((vv[i] - mu) * rstd * g[d] + bta[d]);
  }
}

// ---------------------------------------------------------------------------
// WMMA GEMM: Y[M,N] = epi( X[M,K](bf16) @ W[N,K]^T(f32->bf16) + bias )
// 128x64 block tile, BK=32, 8 waves; each wave owns a 2x2 grid of 16x16 C
// tiles -> 4 WMMAs per k-step on 2 A-frags + 2 B-frags (1 wmma per ds frag).
// ---------------------------------------------------------------------------
__global__ __launch_bounds__(256)
void gemm_wmma(const bf16* __restrict__ X, const float* __restrict__ W,
               const float* __restrict__ bias, const float* __restrict__ residual,
               float* __restrict__ outF, bf16* __restrict__ outB,
               bf16* __restrict__ qp, bf16* __restrict__ kp, bf16* __restrict__ vp,
               int M, int N, int K, int colOffset, int mode) {
  __shared__ bf16 Xs[128 * 32];
  __shared__ bf16 Ws[64 * 32];
  const int tid = threadIdx.x;
  const int lane = tid & 31;
  const int wave = tid >> 5;
  const int ntn = N >> 6;
  const int m0 = (blockIdx.x / ntn) * 128;
  const int n0 = (blockIdx.x % ntn) * 64;
  const int nw = wave & 1;    // two 32-col halves
  const int mw = wave >> 1;   // four 32-row quarters
  const int lhi = (lane < 16) ? 0 : 1;
  const int ln15 = lane & 15;
  const int off = lhi * 8;
  (void)M;

  v8f acc[2][2];
#pragma unroll
  for (int i = 0; i < 2; i++)
#pragma unroll
    for (int j = 0; j < 2; j++) { v8f z = {}; acc[i][j] = z; }

  const int sxr = tid >> 1;          // 0..127 X staging row
  const int sxc = (tid & 1) * 16;    // 0,16
  const int swr = tid >> 2;          // 0..63 W staging row
  const int swc = (tid & 3) * 8;     // 0,8,16,24

  for (int k0 = 0; k0 < K; k0 += 32) {
    __syncthreads();
    { // stage X tile (bf16, 32B per thread)
      const bf16* src = X + (size_t)(m0 + sxr) * K + k0 + sxc;
      if (k0 + 32 < K) __builtin_prefetch(src + 32, 0, 0);
      *(uint4*)&Xs[sxr * 32 + sxc] = *(const uint4*)src;
      *(uint4*)&Xs[sxr * 32 + sxc + 8] = *(const uint4*)(src + 8);
    }
    { // stage W tile with f32 -> bf16 convert
      const float* src = W + (size_t)(n0 + swr) * K + k0 + swc;
      if (k0 + 32 < K) __builtin_prefetch(src + 32, 0, 0);
      float4 f0 = *(const float4*)src;
      float4 f1 = *(const float4*)(src + 4);
      bf16* d = &Ws[swr * 32 + swc];
      d[0] = f2bf(f0.x); d[1] = f2bf(f0.y); d[2] = f2bf(f0.z); d[3] = f2bf(f0.w);
      d[4] = f2bf(f1.x); d[5] = f2bf(f1.y); d[6] = f2bf(f1.z); d[7] = f2bf(f1.w);
    }
    __syncthreads();
    // B fragments (W^T): lane->n, halves->k (lane-half split 16/16)
    v16bf bfrag[2];
#pragma unroll
    for (int j = 0; j < 2; j++)
      bfrag[j] = *(const v16bf*)&Ws[((nw * 2 + j) * 16 + ln15) * 32 + lhi * 16];
    // A fragments: lane->m, halves->k (lane-half split 16/16)
    v16bf afrag[2];
#pragma unroll
    for (int i = 0; i < 2; i++) {
      const bf16* ap = &Xs[((mw * 2 + i) * 16 + ln15) * 32];
      afrag[i] = load2x8(ap + 2 * off, ap + 2 * off + 8);
    }
#pragma unroll
    for (int i = 0; i < 2; i++)
#pragma unroll
      for (int j = 0; j < 2; j++)
        acc[i][j] = wmma_bf16(afrag[i], bfrag[j], acc[i][j]);
  }

  float bv[2]; int ngj[2];
#pragma unroll
  for (int j = 0; j < 2; j++) {
    ngj[j] = n0 + (nw * 2 + j) * 16 + ln15;
    bv[j] = bias ? bias[ngj[j]] : 0.0f;
  }
#pragma unroll
  for (int i = 0; i < 2; i++)
#pragma unroll
    for (int j = 0; j < 2; j++) {
      const int ng = ngj[j];
#pragma unroll
      for (int r = 0; r < 8; r++) {
        const int mloc = (mw * 2 + i) * 16 + r + lhi * 8;
        const size_t mg = (size_t)(m0 + mloc);
        float val = acc[i][j][r] + bv[j];
        if (mode == EP_F32_RES) {
          outF[mg * N + ng] = val + residual[mg * N + ng];
        } else if (mode == EP_BF16) {
          outB[mg * N + ng] = f2bf(val);
        } else if (mode == EP_BF16_RELU) {
          outB[mg * N + ng] = f2bf(val > 0.0f ? val : 0.0f);
        } else { // EP_QKV scatter: [B][H][L][DH] head-major
          const int cg = colOffset + ng;
          const int which = cg >> 9;     // 0=q 1=k 2=v
          const int d = cg & 511;
          const int h = d >> 6, dh = d & 63;
          const int bidx = (int)(mg >> 10);
          const int lseq = (int)(mg & 1023);
          bf16* dst = (which == 0) ? qp : ((which == 1) ? kp : vp);
          dst[(((size_t)bidx * H_ + h) * L_ + lseq) * DH_ + dh] = f2bf(val);
        }
      }
    }
}

// ---------------------------------------------------------------------------
// Flash attention, one wave per (b, h, 16-row Q tile). Q/K/V bf16 [B][H][L][64].
// K chunks are staged with CDNA5 async global->LDS copies (ASYNCcnt);
// V is transposed through registers into LDS for the P*V B-fragments.
// mode: 0=causal, 1=band(w)+causal, 3=none
// ---------------------------------------------------------------------------
#define SM_K 0                    // Klds [32 key][64 dh]
#define SM_VT (32 * 64)           // Vt   [64 dh][32 key]
#define SM_P  (32 * 64 + 64 * 32) // Plds [16 m][32 key]

__global__ __launch_bounds__(32)
void flash_attn_wmma(const bf16* __restrict__ Q, const bf16* __restrict__ Kt,
                     const bf16* __restrict__ V, bf16* __restrict__ O,
                     int mode, int w) {
  __shared__ bf16 smem[32 * 64 + 64 * 32 + 16 * 32];
  bf16* Klds = smem + SM_K;
  bf16* Vt = smem + SM_VT;
  bf16* Plds = smem + SM_P;
  const int bid = blockIdx.x;
  const int qt = bid & 63;
  const int h = (bid >> 6) & 7;
  const int b = bid >> 9;
  const int t0 = qt * 16;
  const int lane = threadIdx.x;
  const int lhi = lane >> 4;
  const int ln = lane & 15;
  const int off = lhi * 8;
  const size_t headBase = ((size_t)b * H_ + h) * L_ * DH_;
  // LDS byte offset of this lane's K-row (static LDS is allocated from 0).
  const unsigned kldsByte = (unsigned)(SM_K + lane * 64) * 2u;

  // Q A-fragments for k = dh 0..31 and 32..63 (held across the whole loop)
  const bf16* qrow = Q + headBase + (size_t)(t0 + ln) * DH_;
  const v16bf qa0 = load2x8(qrow + 2 * off, qrow + 2 * off + 8);
  const v16bf qa1 = load2x8(qrow + 32 + 2 * off, qrow + 40 + 2 * off);

  v8f acc[4];
  float mrow[8], lrow[8];
#pragma unroll
  for (int d = 0; d < 4; d++) { v8f z = {}; acc[d] = z; }
#pragma unroll
  for (int r = 0; r < 8; r++) { mrow[r] = -3.0e38f; lrow[r] = 0.0f; }

  const int cend = (mode == 3) ? L_ : (t0 + 16);
  int cstart = 0;
  if (mode == 1) { int s = t0 - w; if (s < 0) s = 0; cstart = (s >> 5) << 5; }

  for (int c0 = cstart; c0 < cend; c0 += 32) {
    { // stage K chunk with async global->LDS b128 copies (lane <-> key)
      const unsigned long long ga =
          (unsigned long long)(const void*)(Kt + headBase + (size_t)(c0 + lane) * DH_);
#pragma unroll
      for (int j = 0; j < 8; j++) {
        asm volatile("global_load_async_to_lds_b128 %0, %1, off"
                     :: "v"(kldsByte + (unsigned)(j * 16)),
                        "v"(ga + (unsigned long long)(j * 16))
                     : "memory");
      }
    }
    { // stage V chunk transposed through registers (lane <-> key)
      const bf16* vs = V + headBase + (size_t)(c0 + lane) * DH_;
#pragma unroll
      for (int j = 0; j < 8; j++) {
        uint4 pk = *(const uint4*)(vs + j * 8);
        const bf16* pu = (const bf16*)&pk;
#pragma unroll
        for (int e = 0; e < 8; e++) Vt[(j * 8 + e) * 32 + lane] = pu[e];
      }
    }
    asm volatile("s_wait_asynccnt 0" ::: "memory");
    __syncthreads();

    // S = (Q K^T) * 1/8, two 16-key subtiles, two k=32 WMMAs each
    float sv[2][8];
#pragma unroll
    for (int s = 0; s < 2; s++) {
      v16bf bk0 = *(const v16bf*)&Klds[(s * 16 + ln) * 64 + lhi * 16];
      v16bf bk1 = *(const v16bf*)&Klds[(s * 16 + ln) * 64 + 32 + lhi * 16];
      v8f sc = {};
      sc = wmma_bf16(qa0, bk0, sc);
      sc = wmma_bf16(qa1, bk1, sc);
      const int ngk = c0 + s * 16 + ln;
#pragma unroll
      for (int r = 0; r < 8; r++) {
        const int mg = t0 + r + lhi * 8;
        float x = sc[r] * 0.125f;
        if (mode != 3) {
          if (ngk > mg) x = NEGV;
          else if (mode == 1 && (mg - ngk) > w) x = NEGV;
        }
        sv[s][r] = x;
      }
    }

    // online softmax: row max across the 16 lanes of each half-wave
    float rmax[8];
#pragma unroll
    for (int r = 0; r < 8; r++) rmax[r] = fmaxf(sv[0][r], sv[1][r]);
#pragma unroll
    for (int msk = 1; msk < 16; msk <<= 1)
#pragma unroll
      for (int r = 0; r < 8; r++) rmax[r] = fmaxf(rmax[r], __shfl_xor(rmax[r], msk, 32));

    float rsum[8];
#pragma unroll
    for (int r = 0; r < 8; r++) {
      const float mn = fmaxf(mrow[r], rmax[r]);
      const float scl = __expf(mrow[r] - mn);
      mrow[r] = mn;
      lrow[r] *= scl;
#pragma unroll
      for (int d = 0; d < 4; d++) acc[d][r] *= scl;
      rsum[r] = 0.0f;
    }
#pragma unroll
    for (int s = 0; s < 2; s++)
#pragma unroll
      for (int r = 0; r < 8; r++) {
        const float x = sv[s][r];
        const float p = (x < -5.0e8f) ? 0.0f : __expf(x - mrow[r]);
        rsum[r] += p;
        Plds[(r + lhi * 8) * 32 + s * 16 + ln] = f2bf(p);
      }
#pragma unroll
    for (int msk = 1; msk < 16; msk <<= 1)
#pragma unroll
      for (int r = 0; r < 8; r++) rsum[r] += __shfl_xor(rsum[r], msk, 32);
#pragma unroll
    for (int r = 0; r < 8; r++) lrow[r] += rsum[r];
    __syncthreads();

    // O += P V : one A-frag (16x32) x four B-frags (32x16 dh blocks)
    const bf16* pp = &Plds[ln * 32];
    v16bf pa = load2x8(pp + 2 * off, pp + 2 * off + 8);
#pragma unroll
    for (int d = 0; d < 4; d++) {
      v16bf bv = *(const v16bf*)&Vt[(d * 16 + ln) * 32 + lhi * 16];
      acc[d] = wmma_bf16(pa, bv, acc[d]);
    }
    __syncthreads();
  }

  // write O in [B][L][H][DH] (== [B][L][D]) bf16
#pragma unroll
  for (int d = 0; d < 4; d++)
#pragma unroll
    for (int r = 0; r < 8; r++) {
      const int mg = t0 + r + lhi * 8;
      const float o = acc[d][r] / lrow[r];
      O[((size_t)(b * L_ + mg) * H_ + h) * DH_ + d * 16 + ln] = f2bf(o);
    }
}

// ---------------------------------------------------------------------------
// Gate: softmax_e( x[row] @ gate_w[t] + gate_b[t] ) for t=0,1. E=6.
// ---------------------------------------------------------------------------
__global__ __launch_bounds__(128)
void gate_kernel(const float* __restrict__ x, const float* __restrict__ gw,
                 const float* __restrict__ gb, float* __restrict__ gates) {
  __shared__ float red[12][128];
  const int row = blockIdx.x;
  const int t = threadIdx.x;
  const float* xr = x + (size_t)row * D_;
  float a[12];
#pragma unroll
  for (int i = 0; i < 12; i++) a[i] = 0.0f;
  for (int d = t; d < D_; d += 128) {
    const float xv = xr[d];
    const float* g0 = gw + d * E_;
    const float* g1 = gw + D_ * E_ + d * E_;
#pragma unroll
    for (int e = 0; e < E_; e++) { a[e] += xv * g0[e]; a[6 + e] += xv * g1[e]; }
  }
#pragma unroll
  for (int i = 0; i < 12; i++) red[i][t] = a[i];
  __syncthreads();
  for (int o = 64; o > 0; o >>= 1) {
    if (t < o)
#pragma unroll
      for (int i = 0; i < 12; i++) red[i][t] += red[i][t + o];
    __syncthreads();
  }
  if (t < 2) {
    float lg[6]; float mx = -3.0e38f;
#pragma unroll
    for (int e = 0; e < 6; e++) { lg[e] = red[t * 6 + e][0] + gb[t * 6 + e]; mx = fmaxf(mx, lg[e]); }
    float ssum = 0.0f;
#pragma unroll
    for (int e = 0; e < 6; e++) { lg[e] = __expf(lg[e] - mx); ssum += lg[e]; }
    const float inv = 1.0f / ssum;
#pragma unroll
    for (int e = 0; e < 6; e++) gates[((size_t)t * NROW + row) * 6 + e] = lg[e] * inv;
  }
}

// hmix{0,1} (+)= gates[t,row,e] * cur
__global__ __launch_bounds__(256)
void mix_accum(const float* __restrict__ cur, const float* __restrict__ gates,
               float* __restrict__ h0, float* __restrict__ h1, int e, int init) {
  const size_t i = (size_t)blockIdx.x * 256 + threadIdx.x;
  const size_t row = i >> 9;
  const float g0 = gates[row * 6 + e];
  const float g1 = gates[((size_t)NROW + row) * 6 + e];
  const float c = cur[i];
  if (init) { h0[i] = g0 * c; h1[i] = g1 * c; }
  else { h0[i] += g0 * c; h1[i] += g1 * c; }
}

__global__ __launch_bounds__(256)
void cvt_bf16(const float* __restrict__ src, bf16* __restrict__ dst) {
  const size_t i = (size_t)blockIdx.x * 256 + threadIdx.x;
  dst[i] = f2bf(src[i]);
}

// head2: sigmoid( h1[row] . w2 + b2 ), one wave per row, 256-long dot
__global__ __launch_bounds__(32)
void head2_kernel(const bf16* __restrict__ h1, const float* __restrict__ w2,
                  const float* __restrict__ b2, float* __restrict__ out) {
  const int row = blockIdx.x;
  const int lane = threadIdx.x;
  const bf16* hr = h1 + (size_t)row * 256;
  float s = 0.0f;
#pragma unroll
  for (int i = 0; i < 8; i++) { const int k = lane + 32 * i; s += bf2f(hr[k]) * w2[k]; }
#pragma unroll
  for (int m = 16; m >= 1; m >>= 1) s += __shfl_xor(s, m, 32);
  if (lane == 0) out[row] = 1.0f / (1.0f + __expf(-(s + b2[0])));
}

// ---------------------------------------------------------------------------
extern "C" void kernel_launch(void* const* d_in, const int* in_sizes, int n_in,
                              void* d_out, int out_size, void* d_ws, size_t ws_size,
                              hipStream_t stream) {
  (void)in_sizes; (void)n_in; (void)out_size; (void)ws_size;
  const float* x_item = (const float*)d_in[0];
  const float* x_user = (const float*)d_in[1];
  const float* w_in   = (const float*)d_in[2];
  const float* b_in   = (const float*)d_in[3];
  const float* w_out  = (const float*)d_in[4];
  const float* b_out  = (const float*)d_in[5];
  const float* ln_g   = (const float*)d_in[6];
  const float* ln_b   = (const float*)d_in[7];
  const float* cuw    = (const float*)d_in[8];
  const float* cub    = (const float*)d_in[9];
  const float* ciw    = (const float*)d_in[10];
  const float* cib    = (const float*)d_in[11];
  const float* gw     = (const float*)d_in[12];
  const float* gb     = (const float*)d_in[13];
  const float* hw1    = (const float*)d_in[14];
  const float* hb1    = (const float*)d_in[15];
  const float* hw2    = (const float*)d_in[16];
  const float* hb2    = (const float*)d_in[17];
  float* out = (float*)d_out;

  char* ws = (char*)d_ws;
  size_t off = 0;
  auto alloc = [&](size_t bytes) -> char* {
    char* p = ws + off;
    off += (bytes + 255) & ~(size_t)255;
    return p;
  };
  bf16*  t_ln  = (bf16*)alloc((size_t)NROW * D_ * 2);
  bf16*  u_bf  = (bf16*)alloc((size_t)NROW * D_ * 2);
  bf16*  v_bf  = (bf16*)alloc((size_t)NROW * D_ * 2);
  bf16*  qb    = (bf16*)alloc((size_t)NROW * D_ * 2);
  bf16*  kb    = (bf16*)alloc((size_t)NROW * D_ * 2);
  bf16*  vb    = (bf16*)alloc((size_t)NROW * D_ * 2);
  bf16*  attnb = (bf16*)alloc((size_t)NROW * D_ * 2);
  bf16*  h_bf  = (bf16*)alloc((size_t)NROW * D_ * 2);
  bf16*  h1_bf = (bf16*)alloc((size_t)NROW * 256 * 2);
  float* shared= (float*)alloc((size_t)NROW * D_ * 4);
  float* cur   = (float*)alloc((size_t)NROW * D_ * 4);
  float* hmix0 = (float*)alloc((size_t)NROW * D_ * 4);
  float* hmix1 = (float*)alloc((size_t)NROW * D_ * 4);
  float* gates = (float*)alloc((size_t)2 * NROW * E_ * 4);

  const int attnBlocks = B_ * H_ * (L_ / 16);
  auto launch_gemm = [&](const bf16* X, const float* W, const float* bias,
                         const float* res, float* oF, bf16* oB,
                         int N, int colOff, int mode) {
    const int blocks = (NROW / 128) * (N / 64);
    gemm_wmma<<<blocks, 256, 0, stream>>>(X, W, bias, res, oF, oB, qb, kb, vb,
                                          NROW, N, D_, colOff, mode);
  };
  auto self_attn_expert = [&](const float* ln_src, int e, int amode, int w,
                              const float* res, float* dst) {
    ln_bf16_kernel<<<NROW, 128, 0, stream>>>(ln_src, ln_g + e * D_, ln_b + e * D_, t_ln);
    launch_gemm(t_ln, w_in + (size_t)e * 3 * D_ * D_, b_in + e * 3 * D_,
                nullptr, nullptr, nullptr, 3 * D_, 0, EP_QKV);
    flash_attn_wmma<<<attnBlocks, 32, 0, stream>>>(qb, kb, vb, attnb, amode, w);
    launch_gemm(attnb, w_out + (size_t)e * D_ * D_, b_out + e * D_,
                res, dst, nullptr, D_, 0, EP_F32_RES);
  };
  auto mix = [&](const float* src, int e, int init) {
    mix_accum<<<(NROW * D_) / 256, 256, 0, stream>>>(src, gates, hmix0, hmix1, e, init);
  };

  // gates depend only on x_item
  gate_kernel<<<NROW, 128, 0, stream>>>(x_item, gw, gb, gates);

  // expert 0: shared = x_item + SA0(LN(x_item), causal)
  self_attn_expert(x_item, 0, 0, 0, x_item, shared);
  mix(shared, 0, 1);
  // experts 1,2: shared + SA(LN(shared), causal)
  self_attn_expert(shared, 1, 0, 0, shared, cur);  mix(cur, 1, 0);
  self_attn_expert(shared, 2, 0, 0, shared, cur);  mix(cur, 2, 0);

  // expert 3: cross attention, u = LN(x_user)@Wu^T+bu, v = LN(x_item)@Wi^T+bi
  ln_bf16_kernel<<<NROW, 128, 0, stream>>>(x_user, ln_g + 3 * D_, ln_b + 3 * D_, t_ln);
  launch_gemm(t_ln, cuw, cub, nullptr, nullptr, u_bf, D_, 0, EP_BF16);
  ln_bf16_kernel<<<NROW, 128, 0, stream>>>(x_item, ln_g + 3 * D_, ln_b + 3 * D_, t_ln);
  launch_gemm(t_ln, ciw, cib, nullptr, nullptr, v_bf, D_, 0, EP_BF16);
  launch_gemm(u_bf, w_in + (size_t)3 * 3 * D_ * D_, b_in + 3 * 3 * D_,
              nullptr, nullptr, nullptr, D_, 0, EP_QKV);                 // Q from u
  launch_gemm(v_bf, w_in + (size_t)3 * 3 * D_ * D_ + (size_t)D_ * D_,
              b_in + 3 * 3 * D_ + D_,
              nullptr, nullptr, nullptr, 2 * D_, D_, EP_QKV);            // K,V from v
  flash_attn_wmma<<<attnBlocks, 32, 0, stream>>>(qb, kb, vb, attnb, 3, 0);
  launch_gemm(attnb, w_out + (size_t)3 * D_ * D_, b_out + 3 * D_,
              x_item, cur, nullptr, D_, 0, EP_F32_RES);
  mix(cur, 3, 0);

  // experts 4,5: banded local attention on x_item
  self_attn_expert(x_item, 4, 1, 2, x_item, cur);  mix(cur, 4, 0);
  self_attn_expert(x_item, 5, 1, 3, x_item, cur);  mix(cur, 5, 0);

  // heads
  for (int t = 0; t < 2; t++) {
    const float* hm = t ? hmix1 : hmix0;
    cvt_bf16<<<(NROW * D_) / 256, 256, 0, stream>>>(hm, h_bf);
    launch_gemm(h_bf, hw1 + (size_t)t * 256 * D_, hb1 + t * 256,
                nullptr, nullptr, h1_bf, 256, 0, EP_BF16_RELU);
    head2_kernel<<<NROW, 32, 0, stream>>>(h1_bf, hw2 + t * 256, hb2 + t,
                                          out + (size_t)t * NROW);
  }
}
